// _IPEXLlamaAttentionRef_59176059404723
// MI455X (gfx1250) — matrix operand order repaired and checked by reading, not compiled, not run on hardware
//
#include <hip/hip_runtime.h>

// ---------------------------------------------------------------------------
// MI455X (gfx1250) implementation of the Llama attention reference block.
// Compute-bound (~620 GFLOP vs <1GB of traffic at 23.3 TB/s), so everything
// heavy runs on v_wmma_f32_16x16x32_f16 (f16 in, f32 accumulate).
// Pipeline: f32->f16 convert -> QKV GEMMs -> RoPE -> flash attention -> O GEMM
// (+residual, f32 out). Workspace usage: ~192 MiB of d_ws.
//
// CDNA5-specific data movement: global_load_async_to_lds_b128 (ASYNCcnt) for
// LDS staging when the toolchain exposes the builtin, global_prefetch for the
// next GEMM K-tile, wave32 shfl_xor softmax reductions, explicit s_wait_dscnt
// for the in-wave LDS re-layout round-trip.
// ---------------------------------------------------------------------------

typedef __attribute__((ext_vector_type(16))) _Float16 v16h;
typedef __attribute__((ext_vector_type(8)))  float    v8f;

union Frag16 { v16h v; uint4 u[2]; _Float16 h[16]; };

#define HD    128
#define NH    32
#define NKVH  8
#define SEQ   2048
#define HID   4096
#define KVW   (NKVH * HD)   // 1024

#if defined(__AMDGCN__) && __has_builtin(__builtin_amdgcn_global_load_async_to_lds_b128)
#define USE_ASYNC_LDS 1
#else
#define USE_ASYNC_LDS 0
#endif

#if USE_ASYNC_LDS
// Builtin signature (from hipcc diagnostics): param 0 is a pointer to a
// 16-byte int vector in the global (device) address space, param 1 the same
// in LDS address space, then imm offset + cpol.
typedef int v4i_b128 __attribute__((vector_size(16)));
typedef __attribute__((address_space(1))) v4i_b128* g_b128_t;
typedef __attribute__((address_space(3))) v4i_b128* l_b128_t;

__device__ __forceinline__ void async_copy_b128(const void* g, void* l)
{
    __builtin_amdgcn_global_load_async_to_lds_b128((g_b128_t)g, (l_b128_t)l, 0, 0);
}
__device__ __forceinline__ void async_wait_all()
{
#if __has_builtin(__builtin_amdgcn_s_wait_asynccnt)
    __builtin_amdgcn_s_wait_asynccnt(0);
#else
    asm volatile("s_wait_asynccnt 0" ::: "memory");
#endif
}
#endif

// ---------------------------------------------------------------------------
// fp32 -> f16 convert (float4 in, 4xf16 out)
// ---------------------------------------------------------------------------
__global__ void __launch_bounds__(256)
cvt_f32_f16(const float* __restrict__ x, _Float16* __restrict__ y, int n4)
{
    int i = blockIdx.x * 256 + threadIdx.x;
    if (i < n4) {
        float4 f = ((const float4*)x)[i];
        union { _Float16 h[4]; uint2 u; } p;
        p.h[0] = (_Float16)f.x; p.h[1] = (_Float16)f.y;
        p.h[2] = (_Float16)f.z; p.h[3] = (_Float16)f.w;
        ((uint2*)y)[i] = p.u;
    }
}

// ---------------------------------------------------------------------------
// WMMA GEMM: C[M,N] = A[M,K] * B[K,N], f16 operands, f32 accumulate.
// Block tile 128x128 (256 threads = 8 wave32), wave tile 64x32 (4x2 WMMA tiles),
// K-step 32.  A staged into LDS via async DMA (row-major, stride 40h), B staged
// transposed ([n][k], stride 40h) so each lane's B-fragment is one contiguous
// 32B read.  F32OUT=true adds residual and writes f32 (final O-projection).
// ---------------------------------------------------------------------------
template<bool F32OUT>
__global__ void __launch_bounds__(256)
gemm_wmma(const _Float16* __restrict__ A, const _Float16* __restrict__ B,
          _Float16* __restrict__ Ch, float* __restrict__ Cf,
          const float* __restrict__ resid, int M, int N, int K)
{
    (void)M;
    const int BK = 32, LDAS = 40, LDBS = 40;
    __shared__ _Float16 As[128 * 40];
    __shared__ _Float16 Bs[128 * 40];

    int tid  = threadIdx.x;
    int lane = tid & 31;
    int wv   = tid >> 5;
    int wm   = (wv >> 2) * 64;           // wave row offset: 0 / 64
    int wn   = (wv & 3) * 32;            // wave col offset: 0..96
    int bm   = blockIdx.y * 128;
    int bn   = blockIdx.x * 128;
    int mloc   = lane & 15;
    int koff_a = (lane < 16) ? 0 : 16;   // A-fragment k-segment
    int koff_b = (lane < 16) ? 0 : 16;   // B-fragment k-segment
    int rbase  = (lane < 16) ? 0 : 8;    // C/D row split

    v8f acc[4][2];
    #pragma unroll
    for (int i = 0; i < 4; i++)
        #pragma unroll
        for (int j = 0; j < 2; j++)
            #pragma unroll
            for (int r = 0; r < 8; r++) acc[i][j][r] = 0.f;

    int arow = tid >> 1, apart = tid & 1;        // A: 128 rows x 2 uint4
    int bkr  = tid >> 3, bnq  = (tid & 7) * 16;  // B: 32 k-rows x 8 thr x 16 cols

    for (int k0 = 0; k0 < K; k0 += BK) {
        __syncthreads();
        // A tile -> LDS: one 16B DMA (or load/store) per half-row per thread
        const _Float16* asrc = A + (size_t)(bm + arow) * K + k0 + apart * 8;
        _Float16*       adst = As + arow * LDAS + apart * 8;
#if USE_ASYNC_LDS
        async_copy_b128(asrc, adst);
#else
        *(uint4*)adst = *(const uint4*)asrc;
#endif
        // B tile -> LDS transposed to [n][k]
        Frag16 t01;
        const _Float16* bsrc = B + (size_t)(k0 + bkr) * N + bn + bnq;
        t01.u[0] = *(const uint4*)(bsrc);
        t01.u[1] = *(const uint4*)(bsrc + 8);
        #pragma unroll
        for (int i = 0; i < 16; i++) Bs[(bnq + i) * LDBS + bkr] = t01.h[i];
        // prefetch next K-tile while this one is consumed
        if (k0 + BK < K) {
            __builtin_prefetch(asrc + BK, 0, 0);
            __builtin_prefetch(bsrc + (size_t)BK * N, 0, 0);
        }
#if USE_ASYNC_LDS
        async_wait_all();
#endif
        __syncthreads();

        Frag16 a[4], b[2];
        #pragma unroll
        for (int i = 0; i < 4; i++) {
            const _Float16* p = As + (wm + i * 16 + mloc) * LDAS;
            a[i].u[0] = *(const uint4*)(p + koff_a);
            a[i].u[1] = *(const uint4*)(p + 8 + koff_a);
        }
        #pragma unroll
        for (int j = 0; j < 2; j++) {
            const _Float16* p = Bs + (wn + j * 16 + mloc) * LDBS + koff_b;
            b[j].u[0] = *(const uint4*)(p);
            b[j].u[1] = *(const uint4*)(p + 8);
        }
        #pragma unroll
        for (int i = 0; i < 4; i++)
            #pragma unroll
            for (int j = 0; j < 2; j++)
                acc[i][j] = __builtin_amdgcn_wmma_f32_16x16x32_f16(
                    false, a[i].v, false, b[j].v, (short)0, acc[i][j], false, false);
    }

    #pragma unroll
    for (int i = 0; i < 4; i++)
        #pragma unroll
        for (int j = 0; j < 2; j++)
            #pragma unroll
            for (int r = 0; r < 8; r++) {
                int m = bm + wm + i * 16 + rbase + r;
                int n = bn + wn + j * 16 + mloc;
                size_t off = (size_t)m * N + n;
                if constexpr (F32OUT) Cf[off] = acc[i][j][r] + resid[off];
                else                  Ch[off] = (_Float16)acc[i][j][r];
            }
}

// ---------------------------------------------------------------------------
// RoPE (rotate-half), in-place on f16 Q and K, f32 math.
// One thread per (batch*seq, d-pair); loops over heads.
// ---------------------------------------------------------------------------
__global__ void __launch_bounds__(256)
rope_kernel(_Float16* __restrict__ qb, _Float16* __restrict__ kb,
            const int* __restrict__ pos)
{
    int idx = blockIdx.x * 256 + threadIdx.x;   // B*SEQ*64
    int d   = idx & 63;
    int bs  = idx >> 6;
    float p = (float)pos[bs];
    // theta^(-2d/128) = exp(-(2d/128)*ln(10000))
    float inv_freq = __expf(-((float)(2 * d) / 128.0f) * 9.210340371976184f);
    float ang = p * inv_freq;
    float sn, cs;
    __sincosf(ang, &sn, &cs);

    _Float16* q = qb + (size_t)bs * HID + d;
    #pragma unroll
    for (int h = 0; h < NH; h++) {
        float x0 = (float)q[h * HD], x1 = (float)q[h * HD + 64];
        q[h * HD]      = (_Float16)(x0 * cs - x1 * sn);
        q[h * HD + 64] = (_Float16)(x1 * cs + x0 * sn);
    }
    _Float16* k = kb + (size_t)bs * KVW + d;
    #pragma unroll
    for (int h = 0; h < NKVH; h++) {
        float x0 = (float)k[h * HD], x1 = (float)k[h * HD + 64];
        k[h * HD]      = (_Float16)(x0 * cs - x1 * sn);
        k[h * HD + 64] = (_Float16)(x1 * cs + x0 * sn);
    }
}

// ---------------------------------------------------------------------------
// Flash attention (causal, GQA 4:1). Block = 128 threads = 4 waves; each wave
// owns a 16-row q tile with the full d=128 accumulator in registers (8x v8f).
// Per 32-column kv step:
//   - V tile DMA'd to per-wave LDS via async_to_lds (ASYNCcnt), overlapped
//     with the 8 score WMMAs (QK^T over d=128) and the online softmax
//   - K fragments batch-loaded (one clause of 8 b128 loads per score tile)
//   - P re-laid-out D-tile -> A-fragment through LDS (s_wait_dscnt)
//   - 8 PV WMMAs after a single s_wait_asynccnt
// ---------------------------------------------------------------------------
__global__ void __launch_bounds__(128)
flash_kernel(const _Float16* __restrict__ qb, const _Float16* __restrict__ kb,
             const _Float16* __restrict__ vb, _Float16* __restrict__ ob)
{
    const int LDV = 136, LDP = 40;
    __shared__ _Float16 Vlds[4][32 * 136];
    __shared__ _Float16 Plds[4][16 * 40];

    int lane = threadIdx.x & 31, wv = threadIdx.x >> 5;
    int b = blockIdx.x >> 5, h = blockIdx.x & 31, kvh = h >> 2;
    int q0 = blockIdx.y * 64 + wv * 16;
    int mloc   = lane & 15;
    int koff_a = (lane < 16) ? 0 : 16;
    int koff_b = (lane < 16) ? 0 : 16;
    int rbase  = (lane < 16) ? 0 : 8;

    // Q A-fragments for the 4 k=32 chunks of d=128 (kept in registers)
    Frag16 aQ[4];
    const _Float16* qrow = qb + (size_t)(b * SEQ + q0 + mloc) * HID + h * HD;
    #pragma unroll
    for (int c = 0; c < 4; c++) {
        aQ[c].u[0] = *(const uint4*)(qrow + c * 32 + koff_a);
        aQ[c].u[1] = *(const uint4*)(qrow + c * 32 + 8 + koff_a);
    }

    float mi[8], li[8];
    v8f o[8];
    #pragma unroll
    for (int r = 0; r < 8; r++) { mi[r] = -1e30f; li[r] = 0.f; }
    #pragma unroll
    for (int dt = 0; dt < 8; dt++)
        #pragma unroll
        for (int r = 0; r < 8; r++) o[dt][r] = 0.f;

    const float scale = 0.08838834764831845f;   // 1/sqrt(128)
    _Float16* vsh = &Vlds[wv][0];
    _Float16* psh = &Plds[wv][0];

    for (int kv0 = 0; kv0 < q0 + 16; kv0 += 32) {
        // stage V[kv0..kv0+31][0..127] into per-wave LDS (one row per lane).
        // With async DMA this overlaps the score WMMAs + softmax below.
        {
            int kvp = kv0 + lane; if (kvp > SEQ - 1) kvp = SEQ - 1;
            const _Float16* src = vb + (size_t)(b * SEQ + kvp) * KVW + kvh * HD;
            _Float16*       dst = vsh + lane * LDV;
            #pragma unroll
            for (int i = 0; i < 16; i++) {
#if USE_ASYNC_LDS
                async_copy_b128(src + i * 8, dst + i * 8);
#else
                *(uint4*)(dst + i * 8) = *(const uint4*)(src + i * 8);
#endif
            }
        }

        // scores: two 16x16 tiles, contraction over d=128 in 4 WMMAs each.
        // Batch the 8 K-fragment loads per tile into one clause before the
        // WMMA chain so only one loadcnt wait is paid per tile.
        v8f s[2];
        #pragma unroll
        for (int t = 0; t < 2; t++) {
            #pragma unroll
            for (int r = 0; r < 8; r++) s[t][r] = 0.f;
            int ckv = kv0 + t * 16;
            int kvp = ckv + mloc; if (kvp > SEQ - 1) kvp = SEQ - 1;
            const _Float16* krow = kb + (size_t)(b * SEQ + kvp) * KVW + kvh * HD;
            Frag16 bk[4];   // lane = kv column, contiguous 32B of d
            #pragma unroll
            for (int kc = 0; kc < 4; kc++) {
                bk[kc].u[0] = *(const uint4*)(krow + kc * 32 + koff_b);
                bk[kc].u[1] = *(const uint4*)(krow + kc * 32 + koff_b + 8);
            }
            #pragma unroll
            for (int kc = 0; kc < 4; kc++)
                s[t] = __builtin_amdgcn_wmma_f32_16x16x32_f16(
                    false, aQ[kc].v, false, bk[kc].v, (short)0, s[t], false, false);
            #pragma unroll
            for (int r = 0; r < 8; r++) {            // scale + causal mask
                int rr = q0 + rbase + r;
                int cc = ckv + mloc;
                float vsc = s[t][r] * scale;
                s[t][r] = (cc <= rr) ? vsc : -1e30f;
            }
        }

        // online softmax: row stats via 16-lane shfl_xor reductions
        float alpha[8];
        #pragma unroll
        for (int r = 0; r < 8; r++) {
            float mt = fmaxf(s[0][r], s[1][r]);
            #pragma unroll
            for (int off = 8; off >= 1; off >>= 1)
                mt = fmaxf(mt, __shfl_xor(mt, off, 32));
            float nm = fmaxf(mi[r], mt);
            alpha[r] = __expf(mi[r] - nm);
            float p0 = __expf(s[0][r] - nm);
            float p1 = __expf(s[1][r] - nm);
            s[0][r] = p0; s[1][r] = p1;
            float ps = p0 + p1;
            #pragma unroll
            for (int off = 8; off >= 1; off >>= 1)
                ps += __shfl_xor(ps, off, 32);
            li[r] = li[r] * alpha[r] + ps;
            mi[r] = nm;
        }
        #pragma unroll
        for (int dt = 0; dt < 8; dt++)
            #pragma unroll
            for (int r = 0; r < 8; r++) o[dt][r] *= alpha[r];

        // P: D-tile layout -> A-fragment layout via per-wave LDS round-trip
        #pragma unroll
        for (int r = 0; r < 8; r++) {
            psh[(rbase + r) * LDP + mloc]      = (_Float16)s[0][r];
            psh[(rbase + r) * LDP + 16 + mloc] = (_Float16)s[1][r];
        }
        asm volatile("s_wait_dscnt 0" ::: "memory");
        Frag16 aP;
        {
            const _Float16* p = psh + mloc * LDP;
            aP.u[0] = *(const uint4*)(p + koff_a);
            aP.u[1] = *(const uint4*)(p + 8 + koff_a);
        }

#if USE_ASYNC_LDS
        async_wait_all();    // V tile DMA must have landed before the gathers
#endif
        // O += P * V  : 8 d-column tiles
        #pragma unroll
        for (int dt = 0; dt < 8; dt++) {
            Frag16 bv;   // lane = d column, gather 16 kv rows from LDS
            int ncol = dt * 16 + mloc;
            #pragma unroll
            for (int kk = 0; kk < 16; kk++)
                bv.h[kk] = vsh[(koff_b + kk) * LDV + ncol];
            o[dt] = __builtin_amdgcn_wmma_f32_16x16x32_f16(
                false, aP.v, false, bv.v, (short)0, o[dt], false, false);
        }
    }

    // normalize and write f16 attention output [B*S, H*D]
    float inv[8];
    #pragma unroll
    for (int r = 0; r < 8; r++) inv[r] = 1.0f / li[r];
    _Float16* orow = ob + (size_t)(b * SEQ + q0) * HID + h * HD;
    #pragma unroll
    for (int dt = 0; dt < 8; dt++)
        #pragma unroll
        for (int r = 0; r < 8; r++)
            orow[(size_t)(rbase + r) * HID + dt * 16 + mloc] =
                (_Float16)(o[dt][r] * inv[r]);
}

// ---------------------------------------------------------------------------
// host launcher
// ---------------------------------------------------------------------------
extern "C" void kernel_launch(void* const* d_in, const int* in_sizes, int n_in,
                              void* d_out, int out_size, void* d_ws, size_t ws_size,
                              hipStream_t stream)
{
    (void)in_sizes; (void)n_in; (void)out_size; (void)ws_size;

    const float* hidden   = (const float*)d_in[0];
    const float* residual = (const float*)d_in[1];
    /* d_in[2] = attention_mask: pure causal, re-derived in-kernel */
    const int*   pos      = (const int*)d_in[3];
    const float* Wq       = (const float*)d_in[4];
    const float* Wk       = (const float*)d_in[5];
    const float* Wv       = (const float*)d_in[6];
    const float* Wo       = (const float*)d_in[7];
    float*       out      = (float*)d_out;

    const size_t M = 2 * SEQ;           // 4096 rows
    _Float16* p   = (_Float16*)d_ws;    // ~192 MiB of f16 scratch
    _Float16* Xh  = p; p += M * HID;
    _Float16* Wqh = p; p += (size_t)HID * HID;
    _Float16* Wkh = p; p += (size_t)HID * KVW;
    _Float16* Wvh = p; p += (size_t)HID * KVW;
    _Float16* Woh = p; p += (size_t)HID * HID;
    _Float16* Qb  = p; p += M * HID;
    _Float16* Kb  = p; p += M * KVW;
    _Float16* Vb  = p; p += M * KVW;
    _Float16* Ab  = p; p += M * HID;

    auto cvt = [&](const float* src, _Float16* dst, size_t n) {
        int n4 = (int)(n / 4);
        cvt_f32_f16<<<(n4 + 255) / 256, 256, 0, stream>>>(src, dst, n4);
    };
    cvt(hidden, Xh, M * HID);
    cvt(Wq, Wqh, (size_t)HID * HID);
    cvt(Wk, Wkh, (size_t)HID * KVW);
    cvt(Wv, Wvh, (size_t)HID * KVW);
    cvt(Wo, Woh, (size_t)HID * HID);

    // QKV projections (f16 out)
    gemm_wmma<false><<<dim3(HID / 128, M / 128), 256, 0, stream>>>(
        Xh, Wqh, Qb, nullptr, nullptr, (int)M, HID, HID);
    gemm_wmma<false><<<dim3(KVW / 128, M / 128), 256, 0, stream>>>(
        Xh, Wkh, Kb, nullptr, nullptr, (int)M, KVW, HID);
    gemm_wmma<false><<<dim3(KVW / 128, M / 128), 256, 0, stream>>>(
        Xh, Wvh, Vb, nullptr, nullptr, (int)M, KVW, HID);

    // RoPE on Q and K
    rope_kernel<<<(2 * SEQ * 64) / 256, 256, 0, stream>>>(Qb, Kb, pos);

    // causal GQA flash attention
    flash_kernel<<<dim3(2 * NH, SEQ / 64), 128, 0, stream>>>(Qb, Kb, Vb, Ab);

    // O projection + residual (f32 out)
    gemm_wmma<true><<<dim3(HID / 128, M / 128), 256, 0, stream>>>(
        Ab, Woh, nullptr, out, residual, (int)M, HID, HID);
}
